// LoRADoRA_31456340476487
// MI455X (gfx1250) — matrix operand order repaired and checked by reading, not compiled, not run on hardware
//
#include <hip/hip_runtime.h>
#include <hip/hip_bf16.h>

typedef __attribute__((ext_vector_type(16))) __bf16 v16bf;
typedef __attribute__((ext_vector_type(8)))  __bf16 v8bf;
typedef __attribute__((ext_vector_type(8)))  float  v8f;

#define DIM_IN   2048
#define DIM_OUT  2048
#define RANK     16
#define M_TOTAL  (8 * 2048)

#define BM  128
#define BN  256
#define BK  32
#define NITER (DIM_IN / BK)
#define LDT 40   // padded LDS row stride in bf16 elements (80 bytes)

// ---------------------------------------------------------------------------
// Kernel 1: wEff = (W + A*B) * (||W_col|| / ||(W+A*B)_col||), cast to bf16.
// ---------------------------------------------------------------------------
__global__ __launch_bounds__(256) void dora_prep_weff(
    const float* __restrict__ weight, const float* __restrict__ aMat,
    const float* __restrict__ bMat, __bf16* __restrict__ weff)
{
    __shared__ float Bl[RANK][64];
    __shared__ float redW[4][64];
    __shared__ float redL[4][64];
    __shared__ float scaleS[64];

    const int i0 = blockIdx.x * 64;
    const int t  = threadIdx.x;

    for (int idx = t; idx < RANK * 64; idx += 256) {
        int r = idx >> 6, c = idx & 63;
        Bl[r][c] = bMat[r * DIM_IN + i0 + c];
    }
    __syncthreads();

    const int c  = t & 63;
    const int rg = t >> 6;

    float sw = 0.f, sl = 0.f;
    for (int o = rg; o < DIM_OUT; o += 4) {
        const float* ar = aMat + o * RANK;
        float w = weight[(size_t)o * DIM_IN + i0 + c];
        float l = w;
        #pragma unroll
        for (int r = 0; r < RANK; ++r) l += ar[r] * Bl[r][c];
        sw += w * w;
        sl += l * l;
    }
    redW[rg][c] = sw;
    redL[rg][c] = sl;
    __syncthreads();

    if (t < 64) {
        float a = redW[0][t] + redW[1][t] + redW[2][t] + redW[3][t];
        float b = redL[0][t] + redL[1][t] + redL[2][t] + redL[3][t];
        scaleS[t] = __builtin_sqrtf(a / b);
    }
    __syncthreads();

    const float sc = scaleS[c];
    for (int o = rg; o < DIM_OUT; o += 4) {
        const float* ar = aMat + o * RANK;
        float l = weight[(size_t)o * DIM_IN + i0 + c];
        #pragma unroll
        for (int r = 0; r < RANK; ++r) l += ar[r] * Bl[r][c];
        weff[(size_t)o * DIM_IN + i0 + c] = (__bf16)(l * sc);
    }
}

// ---------------------------------------------------------------------------
// Kernel 2: out = X * wEff^T + bias.  Block 128x256, 8 waves, wave tile 64x64.
// B tiles: global_load_async_to_lds_b128 (ASYNCcnt), double-buffered.
// A tiles: reg-pipelined f32 loads -> v_cvt_pk_bf16_f32 -> ds_store_b128.
// ---------------------------------------------------------------------------
__global__ __launch_bounds__(256) void dora_gemm(
    const float* __restrict__ x, const __bf16* __restrict__ weff,
    const float* __restrict__ bias, float* __restrict__ out)
{
    __shared__ __bf16 As[2][BM * LDT];
    __shared__ __bf16 Bs[2][BN * LDT];

    const int t     = threadIdx.x;
    const int lane  = t & 31;
    const int wave  = t >> 5;
    const int half  = lane >> 4;
    const int l16   = lane & 15;
    const int mBlk  = blockIdx.x * BM;
    const int nBlk  = blockIdx.y * BN;
    const int waveM = wave >> 2;   // 0..1 -> 64-row halves
    const int waveN = wave & 3;    // 0..3 -> 64-col slices

    // staging coordinates
    const int aRow = t >> 1;            // 0..127
    const int aCol = (t & 1) * 16;      // 0 or 16 (f32 elements)
    const float*  aPtr = x    + (size_t)(mBlk + aRow) * DIM_IN + aCol;
    const __bf16* bPtr = weff + (size_t)(nBlk + t) * DIM_IN;   // row t of B tile

    v8f acc[4][4];
    #pragma unroll
    for (int i = 0; i < 4; ++i)
        #pragma unroll
        for (int j = 0; j < 4; ++j)
            acc[i][j] = (v8f){0.f,0.f,0.f,0.f,0.f,0.f,0.f,0.f};

    float4 ar[4];

    // ---- prologue: first A slice into regs, first B slice via async DMA ----
    #pragma unroll
    for (int j = 0; j < 4; ++j) ar[j] = *(const float4*)(aPtr + j * 4);
    {
        const unsigned ldsBase =
            (unsigned)(unsigned long long)&Bs[0][t * LDT];
        #pragma unroll
        for (int j = 0; j < 4; ++j) {
            const unsigned ldsA = ldsBase + j * 16;
            const unsigned long long g =
                (unsigned long long)(const void*)(bPtr + j * 8);
            asm volatile("global_load_async_to_lds_b128 %0, %1, off"
                         :: "v"(ldsA), "v"(g) : "memory");
        }
    }

    for (int it = 0; it < NITER; ++it) {
        const int cur = it & 1;

        // ---- commit A regs -> LDS (cvt + 2x ds_store_b128) ----
        alignas(16) __bf16 tmp[16];
        #pragma unroll
        for (int j = 0; j < 4; ++j) {
            tmp[4*j+0] = (__bf16)ar[j].x; tmp[4*j+1] = (__bf16)ar[j].y;
            tmp[4*j+2] = (__bf16)ar[j].z; tmp[4*j+3] = (__bf16)ar[j].w;
        }
        *(v8bf*)&As[cur][aRow * LDT + aCol]     = *(const v8bf*)&tmp[0];
        *(v8bf*)&As[cur][aRow * LDT + aCol + 8] = *(const v8bf*)&tmp[8];

        // B[cur] DMA complete, then workgroup handoff
        asm volatile("s_wait_asynccnt 0x0" ::: "memory");
        __syncthreads();

        // ---- kick off next slice (overlaps with WMMA below) ----
        if (it + 1 < NITER) {
            const int k1 = (it + 1) * BK;
            #pragma unroll
            for (int j = 0; j < 4; ++j)
                ar[j] = *(const float4*)(aPtr + k1 + j * 4);
            const unsigned ldsBase =
                (unsigned)(unsigned long long)&Bs[cur ^ 1][t * LDT];
            #pragma unroll
            for (int j = 0; j < 4; ++j) {
                const unsigned ldsA = ldsBase + j * 16;
                const unsigned long long g =
                    (unsigned long long)(const void*)(bPtr + k1 + j * 8);
                asm volatile("global_load_async_to_lds_b128 %0, %1, off"
                             :: "v"(ldsA), "v"(g) : "memory");
            }
        }

        // ---- fragments (CDNA5 16-bit WMMA swizzle) ----
        v16bf afrag[4];
        #pragma unroll
        for (int mi = 0; mi < 4; ++mi) {
            int row = waveM * 64 + mi * 16 + l16;
            const v8bf lo = *(const v8bf*)&As[cur][row * LDT + half * 8];
            const v8bf hi = *(const v8bf*)&As[cur][row * LDT + 16 + half * 8];
            afrag[mi] = __builtin_shufflevector(lo, hi,
                0,1,2,3,4,5,6,7,8,9,10,11,12,13,14,15);
        }
        v16bf bfrag[4];
        #pragma unroll
        for (int ni = 0; ni < 4; ++ni) {
            int row = waveN * 64 + ni * 16 + l16;
            const v8bf lo = *(const v8bf*)&Bs[cur][row * LDT + half * 8];
            const v8bf hi = *(const v8bf*)&Bs[cur][row * LDT + 16 + half * 8];
            bfrag[ni] = __builtin_shufflevector(lo, hi,
                0,1,2,3,4,5,6,7,8,9,10,11,12,13,14,15);
        }

        // ---- 16 WMMAs per wave per K-step ----
        #pragma unroll
        for (int mi = 0; mi < 4; ++mi)
            #pragma unroll
            for (int ni = 0; ni < 4; ++ni)
                acc[mi][ni] = __builtin_amdgcn_wmma_f32_16x16x32_bf16(
                    false, afrag[mi], false, bfrag[ni],
                    (short)0, acc[mi][ni], false, false);
    }

    // ---- epilogue: C layout = VGPR r -> M=r (lanes 0-15) / M=r+8 (16-31) ----
    #pragma unroll
    for (int mi = 0; mi < 4; ++mi) {
        const int mBase = mBlk + waveM * 64 + mi * 16 + half * 8;
        #pragma unroll
        for (int ni = 0; ni < 4; ++ni) {
            const int n = nBlk + waveN * 64 + ni * 16 + l16;
            const float bv = bias[n];
            #pragma unroll
            for (int r = 0; r < 8; ++r)
                out[(size_t)(mBase + r) * DIM_OUT + n] = acc[mi][ni][r] + bv;
        }
    }
}

// ---------------------------------------------------------------------------
extern "C" void kernel_launch(void* const* d_in, const int* in_sizes, int n_in,
                              void* d_out, int out_size, void* d_ws, size_t ws_size,
                              hipStream_t stream) {
    const float* x      = (const float*)d_in[0];
    const float* weight = (const float*)d_in[1];
    const float* bias   = (const float*)d_in[2];
    const float* aMat   = (const float*)d_in[3];
    const float* bMat   = (const float*)d_in[4];
    __bf16* weff = (__bf16*)d_ws;           // 8 MB scratch
    float*  out  = (float*)d_out;

    dora_prep_weff<<<dim3(DIM_IN / 64), 256, 0, stream>>>(weight, aMat, bMat, weff);
    dora_gemm<<<dim3(M_TOTAL / BM, DIM_OUT / BN), 256, 0, stream>>>(x, weff, bias, out);
}
